// DenseRNNLayer_66692252172678
// MI455X (gfx1250) — matrix operand order repaired and checked by reading, not compiled, not run on hardware
//
#include <hip/hip_runtime.h>
#include <hip/hip_bf16.h>

// ---------------------------------------------------------------------------
// Bidirectional 3-layer GRU (Keras reset_after=True) for MI455X (gfx1250).
//   H=512, U=256, L=3, B=64, T=512, EXTRA=1000, RATE=0.1
// Strategy:
//   * bf16 WMMA (v_wmma_f32_16x16x32_bf16) for both the hoisted input
//     projection GEMM and the per-step recurrent GEMM, f32 accumulate.
//   * GEMM waves are register-blocked 2 M-tiles x 4 N-tiles (32x64 output)
//     to lift L2 arithmetic intensity from ~8 to ~21 FLOP/byte.
//   * Recurrence: one persistent 1024-thread workgroup per direction.
//     h (64x256 f32) and hp (64x768 f32) live in LDS (256 KB of the WGP's
//     320 KB). 192 WMMA tiles per step = 6 tiles/wave * 32 waves, 8 K-steps each.
//   * Dropout: deterministic hash RNG (rate 0.1, scale 1/0.9) gated by the
//     `training` flag read on-device.
// ---------------------------------------------------------------------------

typedef __attribute__((ext_vector_type(16))) __bf16 v16bf;
typedef __attribute__((ext_vector_type(8)))  __bf16 v8bf;
typedef __attribute__((ext_vector_type(8)))  float  v8f;

#define HD   512
#define UD   256
#define G3   768          // 3*U
#define BB   64
#define TT   512
#define MROWS (BB*TT)     // 32768
#define EXTRA 1000

__device__ __forceinline__ __bf16 f2bf(float f) { return (__bf16)f; }

__device__ __forceinline__ v16bf load_afrag(const __bf16* xrow, int k0, int half) {
  // A fragment (16x32 bf16): lane holds row m, K = {k0+half*8 ..+7} U {+16..}
  v8bf alo = *(const v8bf*)(xrow + k0 + half * 8);
  v8bf ahi = *(const v8bf*)(xrow + k0 + 16 + half * 8);
  return __builtin_shufflevector(alo, ahi,
      0, 1, 2, 3, 4, 5, 6, 7, 8, 9, 10, 11, 12, 13, 14, 15);
}

// ---------------------------------------------------------------------------
// f32 -> bf16 elementwise convert (for weight stacks)
// ---------------------------------------------------------------------------
__global__ void k_f32_to_bf16(const float* __restrict__ src,
                              __bf16* __restrict__ dst, long n) {
  long i = (long)blockIdx.x * blockDim.x + threadIdx.x;
  long stride = (long)gridDim.x * blockDim.x;
  for (; i < n; i += stride) dst[i] = f2bf(src[i]);
}

// ---------------------------------------------------------------------------
// states [b*T+t, H] f32  ->  x [t*B+b, H] bf16  (time-major reorder)
// ---------------------------------------------------------------------------
__global__ void k_reorder_states(const float* __restrict__ S,
                                 __bf16* __restrict__ X) {
  long idx = (long)blockIdx.x * blockDim.x + threadIdx.x;   // over 32768*512
  if (idx >= (long)MROWS * HD) return;
  long row = idx >> 9;          // t*64 + b
  long h   = idx & (HD - 1);
  long b   = row & 63;
  long t   = row >> 6;
  X[idx] = f2bf(S[((b << 9) | t) * HD + h]);
}

// ---------------------------------------------------------------------------
// Input projection GEMM:  XP[M,768] = X[M,512] @ W[512,768] + bias.
// Each wave owns a 32x64 output block (2 M-tiles x 4 N-tiles): A fragments
// reused 4x, B fragments reused 2x -> ~21 FLOP/byte from L2.
// Job grid: (2048/2) M-groups x (48/4) N-groups = 12288 wave jobs.
// ---------------------------------------------------------------------------
__global__ void __launch_bounds__(256)
k_gemm_xproj(const __bf16* __restrict__ X,
             const __bf16* __restrict__ W,
             const float*  __restrict__ bias,
             __bf16* __restrict__ XP) {
  const int lane = threadIdx.x & 31;
  const int wave = threadIdx.x >> 5;
  const long job = (long)blockIdx.x * 8 + wave;
  const long numJobs = (long)(MROWS / 32) * (G3 / 64);      // 1024 * 12
  if (job >= numJobs) return;                                // wave-uniform
  const int mg = (int)(job / (G3 / 64));                     // 0..1023
  const int ng = (int)(job % (G3 / 64));                     // 0..11
  const int m    = lane & 15;
  const int half = lane >> 4;

  const __bf16* xrow0 = X + (long)(mg * 32 + m) * HD;
  const __bf16* xrow1 = xrow0 + (long)16 * HD;
  v8f acc[2][4] = {};
  for (int k0 = 0; k0 < HD; k0 += 32) {
    v16bf a0 = load_afrag(xrow0, k0, half);
    v16bf a1 = load_afrag(xrow1, k0, half);
    // B fragment: lane holds N-col lane%16, K = {k0+half*8 ..+7} U {+16..}
    const __bf16* bp = W + (long)k0 * G3 + ng * 64 + m;
#pragma unroll
    for (int j = 0; j < 4; ++j) {
      v16bf b;
#pragma unroll
      for (int e = 0; e < 8; ++e) {
        b[e]     = bp[(long)(half * 8 + e) * G3 + j * 16];
        b[e + 8] = bp[(long)(16 + half * 8 + e) * G3 + j * 16];
      }
      acc[0][j] = __builtin_amdgcn_wmma_f32_16x16x32_bf16(
          false, a0, false, b, (short)0, acc[0][j], false, false);
      acc[1][j] = __builtin_amdgcn_wmma_f32_16x16x32_bf16(
          false, a1, false, b, (short)0, acc[1][j], false, false);
    }
  }
#pragma unroll
  for (int mi = 0; mi < 2; ++mi) {
#pragma unroll
    for (int j = 0; j < 4; ++j) {
      const int n = ng * 64 + j * 16 + m;                    // C col = lane%16
      const float bv = bias[n];
      __bf16* outc = XP + (long)(mg * 32 + mi * 16 + half * 8) * G3 + n;
#pragma unroll
      for (int v = 0; v < 8; ++v) outc[(long)v * G3] = f2bf(acc[mi][j][v] + bv);
    }
  }
}

// ---------------------------------------------------------------------------
// Persistent GRU recurrence. grid = {2} (dir 0 = fwd, dir 1 = bwd),
// block = 1024 threads (32 waves), dynamic LDS = (64*256 + 64*768)*4 = 256 KB.
// Per step: hp = h @ Uw (+br) via 192 WMMA tiles, then gate math, h update.
// ---------------------------------------------------------------------------
__global__ void __launch_bounds__(1024)
k_gru_recurrence(const __bf16* __restrict__ XPf, const __bf16* __restrict__ XPb,
                 const __bf16* __restrict__ Uf,  const __bf16* __restrict__ Ub,
                 const float*  __restrict__ brf, const float*  __restrict__ brb,
                 const unsigned char* __restrict__ mask,
                 float* __restrict__ Y /* [32768, 512] */) {
  extern __shared__ float smem[];
  float* hS  = smem;                 // [64][256]  current hidden state
  float* hpS = smem + BB * UD;       // [64][768]  recurrent pre-activations

  const int dir = blockIdx.x;
  const __bf16* XP = dir ? XPb : XPf;
  const __bf16* Uw = dir ? Ub  : Uf;
  const float*  br = dir ? brb : brf;

  const int tid  = threadIdx.x;
  const int lane = tid & 31;
  const int wave = tid >> 5;
  const int m    = lane & 15;
  const int half = lane >> 4;
  const int tileBase = wave * 6;                 // 6 tiles per wave (192 total)
  const int mt  = tileBase / (G3 / 16);          // shared by all 6 tiles
  const int ntb = tileBase - mt * (G3 / 16);
  const int b_g = tid >> 4;                      // gate row 0..63
  const int u0  = (tid & 15) * 16;               // gate cols u0..u0+15

  for (int i = tid; i < BB * UD; i += 1024) hS[i] = 0.f;
  __syncthreads();

  for (int step = 0; step < TT; ++step) {
    const int t = dir ? (TT - 1 - step) : step;

    // -------- hp = h @ Uw : 6 tiles x 8 WMMA per wave --------
    v8f acc[6] = {};
    const float* hrow = hS + (mt * 16 + m) * UD;
    for (int k0 = 0; k0 < UD; k0 += 32) {
      v16bf a;
#pragma unroll
      for (int e = 0; e < 8; ++e) {
        a[e]     = f2bf(hrow[k0 + half * 8 + e]);
        a[e + 8] = f2bf(hrow[k0 + 16 + half * 8 + e]);
      }
      const __bf16* brow = Uw + (long)k0 * G3 + m;
      __builtin_prefetch(brow + 32L * G3, 0, 0);   // next K-block (L2-resident)
#pragma unroll
      for (int i = 0; i < 6; ++i) {
        v16bf b;
#pragma unroll
        for (int e = 0; e < 8; ++e) {
          b[e]     = brow[(long)(half * 8 + e) * G3 + (ntb + i) * 16];
          b[e + 8] = brow[(long)(16 + half * 8 + e) * G3 + (ntb + i) * 16];
        }
        acc[i] = __builtin_amdgcn_wmma_f32_16x16x32_bf16(
            false, a, false, b, (short)0, acc[i], false, false);
      }
    }
#pragma unroll
    for (int i = 0; i < 6; ++i) {
      const int n = (ntb + i) * 16 + m;
      const float bv = br[n];
      float* dst = hpS + (mt * 16 + half * 8) * G3 + n;
#pragma unroll
      for (int v = 0; v < 8; ++v) dst[(long)v * G3] = acc[i][v] + bv;
    }
    __syncthreads();

    // -------- gate math: each thread owns (b_g, u0..u0+15) --------
    const long xro = (long)(t * BB + b_g) * G3;
    const unsigned char mk = mask[b_g * TT + t];
    float* hptr = hS + b_g * UD;
    const float* hp = hpS + b_g * G3;
    float* yrow = Y + (long)(t * BB + b_g) * HD + dir * UD;
#pragma unroll 4
    for (int j = 0; j < 16; ++j) {
      const int u = u0 + j;
      const float xz = (float)XP[xro + u];
      const float xr = (float)XP[xro + UD + u];
      const float xh = (float)XP[xro + 2 * UD + u];
      const float hz = hp[u], hr = hp[UD + u], hh = hp[2 * UD + u];
      const float hold = hptr[u];
      const float z  = 1.f / (1.f + __expf(-(xz + hz)));
      const float r  = 1.f / (1.f + __expf(-(xr + hr)));
      const float hc = tanhf(xh + r * hh);
      float hn = z * hold + (1.f - z) * hc;
      hn = mk ? hn : hold;                 // Keras mask: carry state & output
      hptr[u] = hn;
      yrow[u] = hn;
    }
    __syncthreads();
  }
}

// ---------------------------------------------------------------------------
// Dropout (+ convert to next layer bf16 x, or reorder into final f32 output)
// ---------------------------------------------------------------------------
__global__ void k_dropout(const float* __restrict__ Yin,
                          const int* __restrict__ training, int layer,
                          __bf16* __restrict__ Xnext, float* __restrict__ Out,
                          int isFinal) {
  long idx = (long)blockIdx.x * blockDim.x + threadIdx.x;
  if (idx >= (long)MROWS * HD) return;
  float v = Yin[idx];
  if (*training) {
    unsigned s = (unsigned)idx * 0x9E3779B1u + (unsigned)(layer + 1) * 0x85EBCA6Bu;
    s ^= s >> 16; s *= 0x7FEB352Du; s ^= s >> 15; s *= 0x846CA68Bu; s ^= s >> 16;
    const float u01 = (float)(s >> 8) * (1.0f / 16777216.0f);
    v = (u01 < 0.1f) ? 0.f : v * (1.f / 0.9f);
  }
  if (isFinal) {
    const long row = idx >> 9, h = idx & (HD - 1);
    const long b = row & 63, t = row >> 6;
    Out[((b << 9) | t) * HD + h] = v;     // back to [b*T+t, H]
  } else {
    Xnext[idx] = f2bf(v);
  }
}

// ---------------------------------------------------------------------------
// Pass-through tail rows: out[B*T:] = states[B*T:]
// ---------------------------------------------------------------------------
__global__ void k_tail_copy(const float* __restrict__ S, float* __restrict__ O) {
  long i = (long)blockIdx.x * blockDim.x + threadIdx.x;   // over 1000*512
  if (i < (long)EXTRA * HD) O[(long)MROWS * HD + i] = S[(long)MROWS * HD + i];
}

// ---------------------------------------------------------------------------
extern "C" void kernel_launch(void* const* d_in, const int* in_sizes, int n_in,
                              void* d_out, int out_size, void* d_ws,
                              size_t ws_size, hipStream_t stream) {
  const float*         states   = (const float*)d_in[0];
  const unsigned char* mask     = (const unsigned char*)d_in[1];
  /* d_in[2] = seq_shape (B,T) — compile-time constants here */
  const int*           training = (const int*)d_in[3];
  const float* Wf = (const float*)d_in[4];
  const float* Uf = (const float*)d_in[5];
  const float* bf = (const float*)d_in[6];
  const float* Wb = (const float*)d_in[7];
  const float* Ub = (const float*)d_in[8];
  const float* bb = (const float*)d_in[9];
  float* out = (float*)d_out;

  // ---- workspace carve-up (~199 MB) ----
  char* ws = (char*)d_ws;
  size_t off = 0;
  auto carve = [&](size_t bytes) -> void* {
    void* p = ws + off;
    off += (bytes + 255) & ~(size_t)255;
    return p;
  };
  __bf16* WfC = (__bf16*)carve((size_t)3 * HD * G3 * 2);
  __bf16* UfC = (__bf16*)carve((size_t)3 * UD * G3 * 2);
  __bf16* WbC = (__bf16*)carve((size_t)3 * HD * G3 * 2);
  __bf16* UbC = (__bf16*)carve((size_t)3 * UD * G3 * 2);
  __bf16* Xb  = (__bf16*)carve((size_t)MROWS * HD * 2);
  __bf16* XPf = (__bf16*)carve((size_t)MROWS * G3 * 2);
  __bf16* XPb = (__bf16*)carve((size_t)MROWS * G3 * 2);
  float*  Y   = (float*) carve((size_t)MROWS * HD * 4);

  // ---- weight stacks -> bf16 ----
  const long nW = (long)3 * HD * G3, nU = (long)3 * UD * G3;
  k_f32_to_bf16<<<2048, 256, 0, stream>>>(Wf, WfC, nW);
  k_f32_to_bf16<<<2048, 256, 0, stream>>>(Uf, UfC, nU);
  k_f32_to_bf16<<<2048, 256, 0, stream>>>(Wb, WbC, nW);
  k_f32_to_bf16<<<2048, 256, 0, stream>>>(Ub, UbC, nU);

  // ---- states -> time-major bf16 x ----
  const long nX = (long)MROWS * HD;
  k_reorder_states<<<(int)((nX + 255) / 256), 256, 0, stream>>>(states, Xb);

  const int gemmGrid = (MROWS / 32) * (G3 / 64) / 8;       // 1536 blocks
  const size_t ldsBytes = (size_t)(BB * UD + BB * G3) * 4; // 256 KB

  for (int l = 0; l < 3; ++l) {
    // hoisted input projections (both directions)
    k_gemm_xproj<<<gemmGrid, 256, 0, stream>>>(
        Xb, WfC + (long)l * HD * G3, bf + (long)l * 2 * G3, XPf);
    k_gemm_xproj<<<gemmGrid, 256, 0, stream>>>(
        Xb, WbC + (long)l * HD * G3, bb + (long)l * 2 * G3, XPb);
    // sequential recurrence, one persistent WG per direction
    k_gru_recurrence<<<2, 1024, ldsBytes, stream>>>(
        XPf, XPb, UfC + (long)l * UD * G3, UbC + (long)l * UD * G3,
        bf + (long)l * 2 * G3 + G3, bb + (long)l * 2 * G3 + G3, mask, Y);
    // dropout; feeds next layer (bf16) or final output (f32, reordered)
    k_dropout<<<(int)((nX + 255) / 256), 256, 0, stream>>>(
        Y, training, l, Xb, out, (l == 2) ? 1 : 0);
  }

  k_tail_copy<<<(int)(((long)EXTRA * HD + 255) / 256), 256, 0, stream>>>(states, out);
}
